// VanillaRNN_15135464751639
// MI455X (gfx1250) — matrix operand order repaired and checked by reading, not compiled, not run on hardware
//
#include <hip/hip_runtime.h>
#include <hip/hip_bf16.h>
#include <math.h>
#include <stdint.h>

typedef __attribute__((ext_vector_type(16))) _Float16     v16h;
typedef __attribute__((ext_vector_type(2)))  _Float16     v2h;
typedef __attribute__((ext_vector_type(8)))  float        v8f;
typedef __attribute__((ext_vector_type(4)))  unsigned int v4u;

// explicit global-address-space pointer so opaque (asm-laundered) pointers
// still lower to global_load_b128 (LOADcnt only), not flat_load (LOADcnt+DScnt)
typedef const v16h __attribute__((address_space(1)))* wgptr;

#define HID     512
#define B_TOT   256
#define B_TILE  16
#define T_LEN   1024
#define NCLS    10
#define KCHUNKS (HID / 32)        // 16 K-chunks of 32 for f16 WMMA
#define NRES    7                 // W fragments pinned in VGPRs across the t-loop
#define NSTR    (KCHUNKS - NRES)  // W fragments streamed from L2 each step (9)
#define PF      3                 // streamed-load pipeline depth (frags in flight)
#define LDS_ROW 260               // dwords per batch column: 256 f16-pairs + 4 pad

// Native CDNA5 transcendental tanh (TRANS32). ISA 7.4: one independent op /
// V_NOP required after a TRANS op before its result is consumed.
__device__ __forceinline__ float fast_tanh(float v) {
    float r;
    asm volatile("v_tanh_f32 %0, %1\n\tv_nop" : "=v"(r) : "v"(v));
    return r;
}

// ---------------------------------------------------------------------------
// Prep: convert W_hh (f32, row-major [512][512]) to f16 and swizzle into the
// exact per-lane WMMA A-fragment order (ISA 7.12.2, 16-bit A 16x32):
//   lane = half*16 + (row%16); VGPR0-3: K = half*8 + {0..7}
//                              VGPR4-7: K = 16 + half*8 + {0..7}
// Fragment (mtile, kc) for lane L is 32 contiguous bytes -> coalesced loads.
// ---------------------------------------------------------------------------
__global__ void rnn_prep_kernel(const float* __restrict__ W_hh,
                                _Float16* __restrict__ Wsw) {
    int e    = blockIdx.x * blockDim.x + threadIdx.x;  // 0 .. 512*512-1
    int j    = e & 15;
    int lane = (e >> 4) & 31;
    int kc   = (e >> 9) & 15;
    int mt   = e >> 13;
    int half = lane >> 4;
    int row  = mt * 16 + (lane & 15);
    int col  = kc * 32 + ((j < 8) ? (half * 8 + j) : (16 + half * 8 + (j - 8)));
    Wsw[e] = (_Float16)W_hh[row * HID + col];
}

// ---------------------------------------------------------------------------
// Persistent recurrence kernel: 16 blocks x 1024 threads (32 waves).
// Block b owns batch columns [16b, 16b+16) for all T steps; h lives in LDS.
// Wave m computes output rows [16m, 16m+16).
// ---------------------------------------------------------------------------
__global__ __launch_bounds__(1024)
void rnn_persistent_kernel(const float* __restrict__ x,
                           const float* __restrict__ h_init,
                           const float* __restrict__ W_hx,
                           const float* __restrict__ b_h,
                           const float* __restrict__ W_ph,
                           const float* __restrict__ b_p,
                           const _Float16* __restrict__ Wsw,
                           float* __restrict__ out) {
    __shared__ unsigned int h_lds[B_TILE * LDS_ROW];   // 16.6 KB

    const int tid  = threadIdx.x;
    const int lane = tid & 31;
    const int mt   = tid >> 5;        // wave id == row tile, 0..31
    const int half = lane >> 4;
    const int n    = lane & 15;       // batch column inside tile (== WMMA N)
    const int col  = blockIdx.x * B_TILE + n;

    // --- init h from h_init (broadcast across batch), packed f16 pairs ---
    for (int d = tid; d < B_TILE * (HID / 2); d += 1024) {
        int nn = d >> 8;              // / 256 pairs
        int p  = d & 255;
        v2h pr = { (_Float16)h_init[2 * p], (_Float16)h_init[2 * p + 1] };
        h_lds[nn * LDS_ROW + p] = __builtin_bit_cast(unsigned int, pr);
    }

    // per-lane epilogue constants for rows m0..m0+7 (C-layout: acc[r] is row
    // mt*16 + 8*half + r, column n). Bias is folded into the WMMA C seed.
    const int m0 = mt * 16 + half * 8;
    float whx[8];
    v8f   bhv;
#pragma unroll
    for (int r = 0; r < 8; ++r) { whx[r] = W_hx[m0 + r]; bhv[r] = b_h[m0 + r]; }

    const float* xcol = x + col * T_LEN;
    // fragment (mt, kc) for this lane lives at wfrag[kc * 32]
    const v16h* wfrag = reinterpret_cast<const v16h*>(Wsw) + mt * KCHUNKS * 32 + lane;

    // pin the first NRES fragments in VGPRs for the whole recurrence
    v16h wres[NRES];
#pragma unroll
    for (int i = 0; i < NRES; ++i) wres[i] = wfrag[i * 32];

    const int bbase = n * LDS_ROW + half * 4;   // dword offset of this lane's B data

    // Loop-carried opaque global-AS pointer: the asm inside the loop redefines
    // it each iteration, defeating LICM of the streamed loads without a
    // per-iteration pointer copy.
    wgptr wp = reinterpret_cast<wgptr>(reinterpret_cast<uintptr_t>(wfrag));

    __syncthreads();

    for (int t = 0; t < T_LEN; ++t) {
        asm volatile("" : "+v"(wp));

        // ---- prime the streamed pipeline: first PF fragments issued now,
        //      landing under the NRES resident WMMAs (~full L2 latency).
        //      sched_barrier pins them here (no sinking to point of use). ----
        v16h sf[PF];
#pragma unroll
        for (int i = 0; i < PF; ++i) sf[i] = wp[(NRES + i) * 32];
        __builtin_amdgcn_sched_barrier(0);

        v8f acc = bhv;   // seed accumulator with b_h (C operand of first WMMA)

        // ---- resident chunks: LDS-only operands, zero load latency ----
#pragma unroll
        for (int kc = 0; kc < NRES; ++kc) {
            union { v4u u[2]; v16h h; } bb;
            const int boff = bbase + kc * 16;
            bb.u[0] = *reinterpret_cast<const v4u*>(&h_lds[boff]);
            bb.u[1] = *reinterpret_cast<const v4u*>(&h_lds[boff + 8]);
            acc = __builtin_amdgcn_wmma_f32_16x16x32_f16(
                      false, wres[kc], false, bb.h, (short)0, acc, false, false);
        }
        // ---- streamed chunks, depth-PF rotating pipeline: consume frag i,
        //      issue frag i+PF. sched_barrier after each prefetch keeps the
        //      loads PF chunks ahead of use -> overlapping live ranges force
        //      distinct VGPRs and partial s_wait_loadcnt instead of drains. ----
#pragma unroll
        for (int i = 0; i < NSTR; ++i) {
            v16h a = sf[i % PF];
            if (i + PF < NSTR) {
                sf[i % PF] = wp[(NRES + i + PF) * 32];
                __builtin_amdgcn_sched_barrier(0);
            }
            union { v4u u[2]; v16h h; } bb;
            const int boff = bbase + (NRES + i) * 16;
            bb.u[0] = *reinterpret_cast<const v4u*>(&h_lds[boff]);
            bb.u[1] = *reinterpret_cast<const v4u*>(&h_lds[boff + 8]);
            acc = __builtin_amdgcn_wmma_f32_16x16x32_f16(
                      false, a, false, bb.h, (short)0, acc, false, false);
        }

        const float xv = xcol[t];

        __syncthreads();  // everyone done reading h_t before overwrite

        // tanh epilogue (native v_tanh_f32), repack rows m0..m0+7 as f16 pairs
        v4u pk;
#pragma unroll
        for (int q = 0; q < 4; ++q) {
            float v0 = fast_tanh(fmaf(whx[2 * q],     xv, acc[2 * q]));
            float v1 = fast_tanh(fmaf(whx[2 * q + 1], xv, acc[2 * q + 1]));
            v2h pr = { (_Float16)v0, (_Float16)v1 };
            pk[q] = __builtin_bit_cast(unsigned int, pr);
        }
        // pairs m0/2 .. m0/2+3 = mt*8 + half*4 + {0..3}: one ds_store_b128
        *reinterpret_cast<v4u*>(&h_lds[n * LDS_ROW + mt * 8 + half * 4]) = pk;

        __syncthreads();  // h_{t+1} visible
    }

    // --- projection: out[col][cls] = W_ph @ h_T + b_p (tiny: 160 lanes) ---
    if (tid < NCLS * B_TILE) {
        const int cls = tid / B_TILE;
        const int nn  = tid % B_TILE;
        float s = 0.f;
        for (int p = 0; p < HID / 2; ++p) {
            v2h hv = __builtin_bit_cast(v2h, h_lds[nn * LDS_ROW + p]);
            s += W_ph[cls * HID + 2 * p]     * (float)hv[0];
            s += W_ph[cls * HID + 2 * p + 1] * (float)hv[1];
        }
        out[(blockIdx.x * B_TILE + nn) * NCLS + cls] = s + b_p[cls];
    }
}

extern "C" void kernel_launch(void* const* d_in, const int* in_sizes, int n_in,
                              void* d_out, int out_size, void* d_ws, size_t ws_size,
                              hipStream_t stream) {
    const float* x      = (const float*)d_in[0];  // [256, 1024]
    const float* h_init = (const float*)d_in[1];  // [512, 1]
    const float* W_hx   = (const float*)d_in[2];  // [512, 1]
    const float* W_hh   = (const float*)d_in[3];  // [512, 512]
    const float* b_h    = (const float*)d_in[4];  // [512, 1]
    const float* W_ph   = (const float*)d_in[5];  // [10, 512]
    const float* b_p    = (const float*)d_in[6];  // [10, 1]

    _Float16* Wsw = (_Float16*)d_ws;              // 512 KB swizzled f16 W_hh

    rnn_prep_kernel<<<(HID * HID) / 256, 256, 0, stream>>>(W_hh, Wsw);
    rnn_persistent_kernel<<<B_TOT / B_TILE, 1024, 0, stream>>>(
        x, h_init, W_hx, b_h, W_ph, b_p, Wsw, (float*)d_out);
}
